// EncoderBlock_58291296142007
// MI455X (gfx1250) — matrix operand order repaired and checked by reading, not compiled, not run on hardware
//
#include <hip/hip_runtime.h>
#include <hip/hip_bf16.h>
#include <stdint.h>
#include <stddef.h>

#define EMBED    1024
#define HEADS    16
#define HEAD_DIM 64
#define FFDIM    4096
#define NBATCH   4
#define SEQ      2048
#define NTOK     (NBATCH * SEQ)

typedef __attribute__((ext_vector_type(16))) __bf16 bf16x16;
typedef __attribute__((ext_vector_type(8)))  float  floatx8;

// ---------------------------------------------------------------- helpers

static __device__ __forceinline__ unsigned short f2bf(float f) {
  union { float f; unsigned u; } v; v.f = f;
  unsigned u = v.u;
  unsigned r = u + 0x7fffu + ((u >> 16) & 1u);   // round-to-nearest-even
  return (unsigned short)(r >> 16);
}

static __device__ __forceinline__ floatx8 zero8() {
  floatx8 z = {0.f, 0.f, 0.f, 0.f, 0.f, 0.f, 0.f, 0.f};
  return z;
}

// Build a 16xbf16 fragment from two 16-byte-aligned halves.
static __device__ __forceinline__ bf16x16 frag_ld(const unsigned short* p0,
                                                  const unsigned short* p1) {
  union { bf16x16 v; uint4 u[2]; } r;
  r.u[0] = *(const uint4*)p0;
  r.u[1] = *(const uint4*)p1;
  return r.v;
}

static __device__ __forceinline__ floatx8 wmma_bf16(bf16x16 a, bf16x16 b, floatx8 c) {
  return __builtin_amdgcn_wmma_f32_16x16x32_bf16(false, a, false, b, (short)0, c,
                                                 false, false);
}

// ---------------------------------------------------------------- convert

__global__ void k_f32_to_bf16(const float* __restrict__ in,
                              unsigned short* __restrict__ out, int n) {
  int i = blockIdx.x * blockDim.x + threadIdx.x;
  int stride = gridDim.x * blockDim.x;
  for (; i < n; i += stride) out[i] = f2bf(in[i]);
}

// ---------------------------------------------------------------- attention
// Grid: (SEQ/64, HEADS, NBATCH), block 128 (4 waves). Wave w owns q rows
// [qblk*64 + 16w, +16) of one (batch, head). Flash-attention over 64-key blocks.

__global__ __launch_bounds__(128) void k_attention(
    const unsigned short* __restrict__ xb,   // [N,S,E] bf16
    const int* __restrict__ mask,            // [N,1,1,S]
    unsigned short* __restrict__ attnb)      // [N,S,E] bf16 out
{
  __shared__ unsigned short Xk [64][72];   // key rows x d   (row-major, padded)
  __shared__ unsigned short XkT[64][72];   // d x key rows   (transposed)
  __shared__ unsigned short Pw [4][16][72]; // per-wave P tile (16 q x 64 keys)

  const int lane = threadIdx.x & 31;
  const int w    = threadIdx.x >> 5;
  const int half = lane >> 4;
  const int l16  = lane & 15;
  const int head = blockIdx.y;
  const int b    = blockIdx.z;
  const int q0   = blockIdx.x * 64 + w * 16;

  // Q fragments (A layout), persistent in registers: rows q0+l16, 64 d values.
  const size_t qbase = ((size_t)(b * SEQ + q0 + l16) * EMBED) + head * HEAD_DIM;
  bf16x16 qa[2];
#pragma unroll
  for (int kc = 0; kc < 2; kc++) {
    const unsigned short* p = xb + qbase + kc * 32;
    qa[kc] = frag_ld(p + half * 8, p + 16 + half * 8);
  }

  floatx8 oacc[4];
  float mrow[8], lrow[8];
#pragma unroll
  for (int t = 0; t < 4; t++) oacc[t] = zero8();
#pragma unroll
  for (int r = 0; r < 8; r++) { mrow[r] = -3.0e38f; lrow[r] = 0.f; }

  const int tid = threadIdx.x;
  const int lr  = tid >> 1;          // 0..63: key row this thread stages
  const int lcb = (tid & 1) * 32;    // column half

  for (int kb = 0; kb < SEQ / 64; kb++) {
    // ---- cooperative stage of the 64x64 bf16 K/V tile (+ transpose) ----
    {
      const unsigned short* src =
          xb + ((size_t)(b * SEQ + kb * 64 + lr) * EMBED) + head * HEAD_DIM + lcb;
      union { uint4 u[4]; unsigned short s[32]; } buf;
#pragma unroll
      for (int i = 0; i < 4; i++) buf.u[i] = ((const uint4*)src)[i];
#pragma unroll
      for (int i = 0; i < 4; i++) *((uint4*)&Xk[lr][lcb + i * 8]) = buf.u[i];
#pragma unroll
      for (int i = 0; i < 32; i++) XkT[lcb + i][lr] = buf.s[i];
    }
    __syncthreads();

    // ---- S = (Q K^T) * 1/sqrt(64), masked ----
    float st[4][8];
#pragma unroll
    for (int nt = 0; nt < 4; nt++) {
      floatx8 s = zero8();
#pragma unroll
      for (int kc = 0; kc < 2; kc++) {
        const unsigned short* p = &Xk[nt * 16 + l16][kc * 32 + half * 8];
        bf16x16 bk = frag_ld(p, p + 16);
        s = wmma_bf16(qa[kc], bk, s);
      }
      const int mk = mask[b * SEQ + kb * 64 + nt * 16 + l16];
#pragma unroll
      for (int r = 0; r < 8; r++)
        st[nt][r] = mk ? s[r] * 0.125f : -1.25e19f;
    }

    // ---- online softmax (row reductions within each 16-lane half) ----
    float corr[8];
#pragma unroll
    for (int r = 0; r < 8; r++) {
      float tm = fmaxf(fmaxf(st[0][r], st[1][r]), fmaxf(st[2][r], st[3][r]));
#pragma unroll
      for (int ms = 1; ms < 16; ms <<= 1) tm = fmaxf(tm, __shfl_xor(tm, ms, 32));
      float mn = fmaxf(mrow[r], tm);
      corr[r]  = __expf(mrow[r] - mn);
      mrow[r]  = mn;
    }
    float psum[8];
#pragma unroll
    for (int r = 0; r < 8; r++) psum[r] = 0.f;
#pragma unroll
    for (int nt = 0; nt < 4; nt++) {
#pragma unroll
      for (int r = 0; r < 8; r++) {
        float p = __expf(st[nt][r] - mrow[r]);
        psum[r] += p;
        Pw[w][r + half * 8][nt * 16 + l16] = f2bf(p);   // C-layout -> LDS
      }
    }
#pragma unroll
    for (int r = 0; r < 8; r++) {
      float rs = psum[r];
#pragma unroll
      for (int ms = 1; ms < 16; ms <<= 1) rs += __shfl_xor(rs, ms, 32);
      lrow[r] = lrow[r] * corr[r] + rs;
#pragma unroll
      for (int nt = 0; nt < 4; nt++) oacc[nt][r] *= corr[r];
    }

    // ---- O += P * V  (P reloaded from LDS in A layout; V from XkT) ----
#pragma unroll
    for (int kc = 0; kc < 2; kc++) {
      const unsigned short* pp = &Pw[w][l16][kc * 32];
      bf16x16 pa = frag_ld(pp + half * 8, pp + 16 + half * 8);
#pragma unroll
      for (int nt = 0; nt < 4; nt++) {
        const unsigned short* vp = &XkT[nt * 16 + l16][kc * 32 + half * 8];
        bf16x16 vb = frag_ld(vp, vp + 16);
        oacc[nt] = wmma_bf16(pa, vb, oacc[nt]);
      }
    }
    __syncthreads();
  }

  // ---- normalize and store bf16 attention output ----
#pragma unroll
  for (int nt = 0; nt < 4; nt++) {
#pragma unroll
    for (int r = 0; r < 8; r++) {
      const int m = r + half * 8;
      const float v = oacc[nt][r] / lrow[r];
      const size_t idx =
          ((size_t)(b * SEQ + q0 + m) * EMBED) + head * HEAD_DIM + nt * 16 + l16;
      attnb[idx] = f2bf(v);
    }
  }
}

// ---------------------------------------------------------------- GEMM
// C[t, j] = sum_k A[t,k] * W[j,k] + bias[j]  (+ReLU) (+residual)
// Block 256 (8 waves) computes 128 rows x 64 cols. W staged in LDS in 128-K slices.

template <bool RELU, bool RES, bool OUTBF>
__global__ __launch_bounds__(256) void k_gemm(
    const unsigned short* __restrict__ A,    // [M, K] bf16
    const unsigned short* __restrict__ W,    // [NJ, K] bf16 (row-major)
    const float* __restrict__ bias,          // [NJ]
    const float* __restrict__ res,           // [M, NJ] fp32 or null
    float* __restrict__ outf,                // [M, NJ] fp32 or null
    unsigned short* __restrict__ outb,       // [M, NJ] bf16 or null
    int K, int NJ)
{
  __shared__ unsigned short Ws[64][136];

  const int lane = threadIdx.x & 31;
  const int w    = threadIdx.x >> 5;
  const int half = lane >> 4;
  const int l16  = lane & 15;
  const int j0   = blockIdx.x * 64;
  const int r0   = blockIdx.y * 128 + w * 16;

  floatx8 acc[4];
#pragma unroll
  for (int t = 0; t < 4; t++) acc[t] = zero8();

  const int tid = threadIdx.x;
  const int wr  = tid >> 2;          // 0..63
  const int wcb = (tid & 3) * 32;    // 0,32,64,96

  for (int ks = 0; ks < K; ks += 128) {
    const unsigned short* src = W + (size_t)(j0 + wr) * K + ks + wcb;
#pragma unroll
    for (int i = 0; i < 4; i++) *((uint4*)&Ws[wr][wcb + i * 8]) = ((const uint4*)src)[i];
    if (ks + 128 < K) __builtin_prefetch(src + 128, 0, 0);
    __syncthreads();

#pragma unroll
    for (int kc = 0; kc < 4; kc++) {
      const size_t ab = (size_t)(r0 + l16) * K + ks + kc * 32;
      bf16x16 af = frag_ld(A + ab + half * 8, A + ab + 16 + half * 8);
#pragma unroll
      for (int nt = 0; nt < 4; nt++) {
        const unsigned short* bp = &Ws[nt * 16 + l16][kc * 32 + half * 8];
        bf16x16 bfrag = frag_ld(bp, bp + 16);
        acc[nt] = wmma_bf16(af, bfrag, acc[nt]);
      }
    }
    __syncthreads();
  }

#pragma unroll
  for (int nt = 0; nt < 4; nt++) {
    const int j = j0 + nt * 16 + l16;
    const float bv = bias[j];
#pragma unroll
    for (int r = 0; r < 8; r++) {
      const int t = r0 + r + half * 8;
      float v = acc[nt][r] + bv;
      if (RELU) v = fmaxf(v, 0.f);
      const size_t idx = (size_t)t * NJ + j;
      if (RES) v += res[idx];
      if (OUTBF) outb[idx] = f2bf(v);
      else       outf[idx] = v;
    }
  }
}

// ---------------------------------------------------------------- LayerNorm
// One block per token row of 1024; fused *(1-p); optional bf16 sidecar output.

__global__ __launch_bounds__(256) void k_layernorm(
    const float* __restrict__ in, const float* __restrict__ gamma,
    const float* __restrict__ beta, float* __restrict__ outf,
    unsigned short* __restrict__ outb)
{
  __shared__ float red[256];
  const int row = blockIdx.x;
  const int tid = threadIdx.x;
  const float* p = in + (size_t)row * EMBED;
  float4 v = ((const float4*)p)[tid];

  red[tid] = v.x + v.y + v.z + v.w;
  __syncthreads();
  for (int o = 128; o > 0; o >>= 1) {
    if (tid < o) red[tid] += red[tid + o];
    __syncthreads();
  }
  const float mean = red[0] * (1.f / EMBED);
  __syncthreads();

  const float dx = v.x - mean, dy = v.y - mean, dz = v.z - mean, dw = v.w - mean;
  red[tid] = dx * dx + dy * dy + dz * dz + dw * dw;
  __syncthreads();
  for (int o = 128; o > 0; o >>= 1) {
    if (tid < o) red[tid] += red[tid + o];
    __syncthreads();
  }
  const float inv = rsqrtf(red[0] * (1.f / EMBED) + 1e-6f);

  const float4 g  = ((const float4*)gamma)[tid];
  const float4 be = ((const float4*)beta)[tid];
  float4 o4;
  o4.x = (g.x * dx * inv + be.x) * 0.9f;
  o4.y = (g.y * dy * inv + be.y) * 0.9f;
  o4.z = (g.z * dz * inv + be.z) * 0.9f;
  o4.w = (g.w * dw * inv + be.w) * 0.9f;
  ((float4*)(outf + (size_t)row * EMBED))[tid] = o4;
  if (outb) {
    unsigned short* ob = outb + (size_t)row * EMBED + tid * 4;
    ob[0] = f2bf(o4.x); ob[1] = f2bf(o4.y);
    ob[2] = f2bf(o4.z); ob[3] = f2bf(o4.w);
  }
}

// ---------------------------------------------------------------- launch

extern "C" void kernel_launch(void* const* d_in, const int* in_sizes, int n_in,
                              void* d_out, int out_size, void* d_ws, size_t ws_size,
                              hipStream_t stream) {
  (void)in_sizes; (void)n_in; (void)out_size; (void)ws_size;

  const float* x    = (const float*)d_in[0];
  const int*   mask = (const int*)d_in[1];
  const float* fc_w = (const float*)d_in[2];
  const float* fc_b = (const float*)d_in[3];
  const float* w1   = (const float*)d_in[4];
  const float* b1   = (const float*)d_in[5];
  const float* w2   = (const float*)d_in[6];
  const float* b2   = (const float*)d_in[7];
  const float* g1   = (const float*)d_in[8];
  const float* be1  = (const float*)d_in[9];
  const float* g2   = (const float*)d_in[10];
  const float* be2  = (const float*)d_in[11];
  float* out = (float*)d_out;

  char* ws = (char*)d_ws;
  size_t off = 0;
  auto alloc = [&](size_t bytes) -> void* {
    void* p = ws + off;
    off += (bytes + 255) & ~(size_t)255;
    return p;
  };

  unsigned short* xb    = (unsigned short*)alloc((size_t)NTOK * EMBED * 2);
  unsigned short* fcwb  = (unsigned short*)alloc((size_t)EMBED * EMBED * 2);
  unsigned short* w1b   = (unsigned short*)alloc((size_t)FFDIM * EMBED * 2);
  unsigned short* w2b   = (unsigned short*)alloc((size_t)EMBED * FFDIM * 2);
  unsigned short* attnb = (unsigned short*)alloc((size_t)NTOK * EMBED * 2);
  float*          t0    = (float*)alloc((size_t)NTOK * EMBED * 4);   // reused
  float*          hbuf  = (float*)alloc((size_t)NTOK * EMBED * 4);
  unsigned short* hb    = (unsigned short*)alloc((size_t)NTOK * EMBED * 2);
  unsigned short* midb  = (unsigned short*)alloc((size_t)NTOK * FFDIM * 2);

  // 1) pack to bf16
  k_f32_to_bf16<<<4096, 256, 0, stream>>>(x,    xb,   NTOK * EMBED);
  k_f32_to_bf16<<<1024, 256, 0, stream>>>(fc_w, fcwb, EMBED * EMBED);
  k_f32_to_bf16<<<4096, 256, 0, stream>>>(w1,   w1b,  FFDIM * EMBED);
  k_f32_to_bf16<<<4096, 256, 0, stream>>>(w2,   w2b,  EMBED * FFDIM);

  // 2) flash attention (Q=K=V=x head-sliced)
  k_attention<<<dim3(SEQ / 64, HEADS, NBATCH), 128, 0, stream>>>(xb, mask, attnb);

  // 3) fc projection + residual(x)  -> t0 fp32
  k_gemm<false, true, false><<<dim3(EMBED / 64, NTOK / 128), 256, 0, stream>>>(
      attnb, fcwb, fc_b, x, t0, nullptr, EMBED, EMBED);

  // 4) LayerNorm1 * 0.9 -> h fp32 + bf16
  k_layernorm<<<NTOK, 256, 0, stream>>>(t0, g1, be1, hbuf, hb);

  // 5) FFN up + ReLU -> midb bf16
  k_gemm<true, false, true><<<dim3(FFDIM / 64, NTOK / 128), 256, 0, stream>>>(
      hb, w1b, b1, nullptr, nullptr, midb, EMBED, FFDIM);

  // 6) FFN down + residual(h) -> t0 fp32 (reused)
  k_gemm<false, true, false><<<dim3(EMBED / 64, NTOK / 128), 256, 0, stream>>>(
      midb, w2b, b2, hbuf, t0, nullptr, FFDIM, EMBED);

  // 7) LayerNorm2 * 0.9 -> final output
  k_layernorm<<<NTOK, 256, 0, stream>>>(t0, g2, be2, out, nullptr);
}
